// MultiHeadAttentionLayer_49701361549706
// MI455X (gfx1250) — matrix operand order repaired and checked by reading, not compiled, not run on hardware
//
#include <hip/hip_runtime.h>
#include <hip/hip_bf16.h>

// MI455X / gfx1250, wave32. WMMA f32_16x16x32_f16 for every matmul.
// Pipeline:
//   0) cvt_kernel x7     : fp32 -> f16 (q,k,v activations + 4 weight matrices), one pass
//   1) gemm_kernel<false> x3 : Q16[b,h,s,d] (pre-scaled 1/8), K16[b,h,s,d], Vt16[b,h,d,s]
//      - async global->LDS DMA staging (ASYNCcnt), double-buffered, 64x64 wave tiles
//   2) attn_kernel       : flash attention, causal mask analytic, 64-key tiles, ctx16[b,s,h*d]
//   3) gemm_kernel<true> : out = ctx16 @ Wo16^T + bo (f32)

#define HIDDEN 1024
#define HEADS  16
#define HEADD  64
#define BATCH  2
#define SLEN   2048
#define MTOK   (BATCH * SLEN)   // 4096 tokens
#define KDIM   1024
#define NDIM   1024
#define LSTR   40               // GEMM LDS row stride (halfs): 80B, breaks 64-bank conflicts
#define ASTR   72               // attention P-slab row stride (halfs)

typedef __attribute__((ext_vector_type(16))) _Float16 v16h;
typedef __attribute__((ext_vector_type(8)))  _Float16 v8h;
typedef __attribute__((ext_vector_type(8)))  float    v8f;

__device__ __forceinline__ v8f wmma16(v16h a, v16h b, v8f c) {
  return __builtin_amdgcn_wmma_f32_16x16x32_f16(false, a, false, b, (short)0, c, false, false);
}

// Async DMA: 16B global -> LDS per lane, tracked by ASYNCcnt (CDNA5 path).
// LDS address operand = 32-bit byte offset (low 32 bits of generic pointer).
__device__ __forceinline__ void async_b128(unsigned lds_off, const void* gaddr) {
  asm volatile("global_load_async_to_lds_b128 %0, %1, off"
               :: "v"(lds_off), "v"(gaddr) : "memory");
}
__device__ __forceinline__ void wait_async0() {
  asm volatile("s_wait_asynccnt 0x0" ::: "memory");
}

// 16x32 f16 A fragment from LDS (row-major, stride `str`).
// lane L: row M=L&15; lanes 0-15 K=0-7/16-23, lanes 16-31 K=8-15/24-31.
__device__ __forceinline__ v16h frag_a_lds(const _Float16* base, int m0, int lane, int str) {
  const _Float16* p = base + (m0 + (lane & 15)) * str + ((lane >> 4) << 3);
  v16h r;
  ((v8h*)&r)[0] = *(const v8h*)p;
  ((v8h*)&r)[1] = *(const v8h*)(p + 16);
  return r;
}
// 32x16 f16 B fragment (B[k][n] = W[n][k]): lane L column N=L&15; lanes 0-15 K=0-7/16-23, lanes 16-31 K=8-15/24-31.
__device__ __forceinline__ v16h frag_b_lds(const _Float16* base, int n0, int lane) {
  const _Float16* p = base + (n0 + (lane & 15)) * LSTR + ((lane >> 4) << 3);
  v16h r;
  ((v8h*)&r)[0] = *(const v8h*)p;
  ((v8h*)&r)[1] = *(const v8h*)(p + 16);
  return r;
}

// -------------------- fp32 -> f16 one-pass convert --------------------
__global__ __launch_bounds__(256)
void cvt_kernel(const float* __restrict__ src, _Float16* __restrict__ dst, int n) {
  int i = (blockIdx.x * 256 + threadIdx.x) * 8;
  if (i >= n) return;
  const float4* s = (const float4*)(src + i);
  float4 a = s[0], b = s[1];
  alignas(16) _Float16 h[8] = {
    (_Float16)a.x, (_Float16)a.y, (_Float16)a.z, (_Float16)a.w,
    (_Float16)b.x, (_Float16)b.y, (_Float16)b.z, (_Float16)b.w };
  *(v8h*)(dst + i) = *(const v8h*)h;
}

// -------------------- GEMM: C[M,N] = A[M,K] @ W[N,K]^T (all f16 in) ------
// 128 threads = 4 waves (2x2); WG tile 128x128; wave tile 64x64; k-step 32.
// Double-buffered async global->LDS staging.
template <bool FINAL>
__global__ __launch_bounds__(128)
void gemm_kernel(const _Float16* __restrict__ A, const _Float16* __restrict__ W,
                 const float* __restrict__ bias, void* __restrict__ dst,
                 float scale, int transpose) {
  __shared__ _Float16 lds_a[2][128 * LSTR];
  __shared__ _Float16 lds_b[2][128 * LSTR];

  const int tid  = threadIdx.x;           // 0..127 (also: staged tile row)
  const int wid  = tid >> 5;
  const int lane = tid & 31;
  const int wm   = wid >> 1;              // 0..1 -> 64-row band
  const int wn   = wid & 1;               // 0..1 -> 64-col band
  const int n0   = blockIdx.x * 128;
  const int m0   = blockIdx.y * 128;

  const unsigned la[2] = { (unsigned)(uintptr_t)&lds_a[0][0],
                           (unsigned)(uintptr_t)&lds_a[1][0] };
  const unsigned lb[2] = { (unsigned)(uintptr_t)&lds_b[0][0],
                           (unsigned)(uintptr_t)&lds_b[1][0] };
  const unsigned ldsrow = (unsigned)(tid * LSTR) * 2u;

  const _Float16* arow = A + (size_t)(m0 + tid) * KDIM;
  const _Float16* wrow = W + (size_t)(n0 + tid) * KDIM;

  v8f acc[4][4];
  #pragma unroll
  for (int i = 0; i < 4; ++i)
    #pragma unroll
    for (int j = 0; j < 4; ++j) acc[i][j] = (v8f){};

  // preload k-tile 0 into buffer 0 (each thread DMAs one 64B row of each tile)
  #pragma unroll
  for (int c = 0; c < 4; ++c) {
    async_b128(la[0] + ldsrow + c * 16, arow + c * 8);
    async_b128(lb[0] + ldsrow + c * 16, wrow + c * 8);
  }
  wait_async0();
  __syncthreads();

  for (int kt = 0; kt < KDIM; kt += 32) {
    const int cur = (kt >> 5) & 1, nxt = cur ^ 1;
    if (kt + 32 < KDIM) {   // DMA next k-tile while computing on current
      #pragma unroll
      for (int c = 0; c < 4; ++c) {
        async_b128(la[nxt] + ldsrow + c * 16, arow + kt + 32 + c * 8);
        async_b128(lb[nxt] + ldsrow + c * 16, wrow + kt + 32 + c * 8);
      }
    }
    v16h af[4], bf[4];
    #pragma unroll
    for (int i = 0; i < 4; ++i) af[i] = frag_a_lds(lds_a[cur], wm * 64 + i * 16, lane, LSTR);
    #pragma unroll
    for (int j = 0; j < 4; ++j) bf[j] = frag_b_lds(lds_b[cur], wn * 64 + j * 16, lane);
    #pragma unroll
    for (int i = 0; i < 4; ++i)
      #pragma unroll
      for (int j = 0; j < 4; ++j) acc[i][j] = wmma16(af[i], bf[j], acc[i][j]);
    if (kt + 32 < KDIM) {
      wait_async0();
      __syncthreads();
    }
  }

  // ---- store ----
  const int rowbase = (lane >> 4) * 8;    // C layout: VGPR r -> row r (+8 for lanes 16-31)
  const int col     = lane & 15;
  if (FINAL) {
    float* out = (float*)dst;
    float bv[4];
    #pragma unroll
    for (int j = 0; j < 4; ++j) bv[j] = bias[n0 + wn * 64 + j * 16 + col];
    #pragma unroll
    for (int i = 0; i < 4; ++i)
      #pragma unroll
      for (int j = 0; j < 4; ++j)
        #pragma unroll
        for (int r = 0; r < 8; ++r) {
          int gm = m0 + wm * 64 + i * 16 + rowbase + r;
          int gn = n0 + wn * 64 + j * 16 + col;
          out[(size_t)gm * NDIM + gn] = acc[i][j][r] + bv[j];
        }
  } else {
    _Float16* out = (_Float16*)dst;
    #pragma unroll
    for (int i = 0; i < 4; ++i)
      #pragma unroll
      for (int j = 0; j < 4; ++j)
        #pragma unroll
        for (int r = 0; r < 8; ++r) {
          int gm = m0 + wm * 64 + i * 16 + rowbase + r;
          int gn = n0 + wn * 64 + j * 16 + col;
          int bb = gm >> 11, s = gm & (SLEN - 1);
          int hh = gn >> 6,  d = gn & (HEADD - 1);
          size_t off = transpose
              ? (((size_t)(bb * HEADS + hh) * HEADD + d) * SLEN + s)   // Vt[b,h,d,s]
              : (((size_t)(bb * HEADS + hh) * SLEN + s) * HEADD + d);  // Q/K[b,h,s,d]
          out[off] = (_Float16)(acc[i][j][r] * scale);
        }
  }
}

// -------------------- Flash attention (causal), 64-key tiles --------------------
// One wave per (b, h, 16-query block). Q pre-scaled by 1/sqrt(d).
__global__ __launch_bounds__(256)
void attn_kernel(const _Float16* __restrict__ Q, const _Float16* __restrict__ K,
                 const _Float16* __restrict__ Vt, _Float16* __restrict__ ctx) {
  __shared__ _Float16 plds_all[8 * 16 * ASTR];   // per-wave P slab (C-layout -> A-layout)
  const int tid  = threadIdx.x;
  const int wid  = tid >> 5;
  const int lane = tid & 31;
  _Float16* plds = plds_all + wid * 16 * ASTR;

  const int gw    = blockIdx.x * 8 + wid;        // 0..4095
  const int qb    = gw & 127;
  const int h     = (gw >> 7) & 15;
  const int b     = gw >> 11;
  const int qbase = qb * 16;
  const int qend  = qbase + 15;

  const size_t headoff = (size_t)(b * HEADS + h) * SLEN * HEADD;
  const int row     = lane & 15;
  const int kb8     = (lane >> 4) * 8;
  const int rowbase = kb8;
  const int col     = lane & 15;

  // Q A-fragments (16x64 -> two 16x32 frags), direct from global
  const _Float16* qp = Q + headoff + (size_t)(qbase + row) * HEADD;
  v16h qa0, qa1;
  ((v8h*)&qa0)[0] = *(const v8h*)(qp + kb8);
  ((v8h*)&qa0)[1] = *(const v8h*)(qp + kb8 + 16);
  ((v8h*)&qa1)[0] = *(const v8h*)(qp + 32 + kb8);
  ((v8h*)&qa1)[1] = *(const v8h*)(qp + 32 + kb8 + 16);

  v8f o[4];
  #pragma unroll
  for (int j = 0; j < 4; ++j) o[j] = (v8f){};
  float mrun[8], lrun[8];
  #pragma unroll
  for (int r = 0; r < 8; ++r) { mrun[r] = -1e30f; lrun[r] = 0.0f; }

  for (int t = 0; t <= qend; t += 64) {
    if (t + 64 <= qend) {   // global_prefetch_b8 for next K/V tiles
      __builtin_prefetch(K + headoff + (size_t)(t + 64) * HEADD, 0, 1);
      __builtin_prefetch(Vt + headoff + t + 64, 0, 1);
    }
    // ---- scores: four 16-key subtiles, 2 WMMAs each over d=64 ----
    v8f c[4];
    #pragma unroll
    for (int x = 0; x < 4; ++x) {
      c[x] = (v8f){};
      if (t + 16 * x <= qend) {            // uniform guard per wave
        int key = t + 16 * x + col;
        if (key > SLEN - 1) key = SLEN - 1;                     // clamp (masked anyway)
        const _Float16* kp = K + headoff + (size_t)key * HEADD + ((lane >> 4) << 3);
        v16h kf0, kf1;
        ((v8h*)&kf0)[0] = *(const v8h*)kp;
        ((v8h*)&kf0)[1] = *(const v8h*)(kp + 16);
        ((v8h*)&kf1)[0] = *(const v8h*)(kp + 32);
        ((v8h*)&kf1)[1] = *(const v8h*)(kp + 48);
        c[x] = wmma16(qa0, kf0, c[x]);
        c[x] = wmma16(qa1, kf1, c[x]);
      }
    }
    // ---- causal mask + online softmax over 64 keys ----
    float sv[4][8];
    #pragma unroll
    for (int x = 0; x < 4; ++x)
      #pragma unroll
      for (int r = 0; r < 8; ++r) {
        int q = qbase + rowbase + r;
        sv[x][r] = (t + 16 * x + col <= q) ? c[x][r] : -1e30f;
      }
    float tm[8];
    #pragma unroll
    for (int r = 0; r < 8; ++r)
      tm[r] = fmaxf(fmaxf(sv[0][r], sv[1][r]), fmaxf(sv[2][r], sv[3][r]));
    #pragma unroll
    for (int off = 1; off < 16; off <<= 1)   // row groups live inside 16-lane halves
      #pragma unroll
      for (int r = 0; r < 8; ++r) tm[r] = fmaxf(tm[r], __shfl_xor(tm[r], off, 32));

    float alpha[8], p[4][8], rs[8];
    #pragma unroll
    for (int r = 0; r < 8; ++r) {
      float mn = fmaxf(mrun[r], tm[r]);
      alpha[r] = __expf(mrun[r] - mn);
      mrun[r]  = mn;
      float acc = 0.0f;
      #pragma unroll
      for (int x = 0; x < 4; ++x) { p[x][r] = __expf(sv[x][r] - mn); acc += p[x][r]; }
      rs[r] = acc;
    }
    #pragma unroll
    for (int off = 1; off < 16; off <<= 1)
      #pragma unroll
      for (int r = 0; r < 8; ++r) rs[r] += __shfl_xor(rs[r], off, 32);
    #pragma unroll
    for (int r = 0; r < 8; ++r) lrun[r] = lrun[r] * alpha[r] + rs[r];
    #pragma unroll
    for (int j = 0; j < 4; ++j)
      #pragma unroll
      for (int r = 0; r < 8; ++r) o[j][r] *= alpha[r];

    // ---- P: C-layout -> A-layout via per-wave LDS slab (16 x 64) ----
    #pragma unroll
    for (int x = 0; x < 4; ++x)
      #pragma unroll
      for (int r = 0; r < 8; ++r)
        plds[(rowbase + r) * ASTR + x * 16 + col] = (_Float16)p[x][r];
    asm volatile("s_wait_dscnt 0x0" ::: "memory");  // wave-local LDS RAW

    v16h pa0, pa1;
    {
      const _Float16* pp = plds + row * ASTR + ((lane >> 4) << 4);
      ((v8h*)&pa0)[0] = *(const v8h*)pp;
      ((v8h*)&pa0)[1] = *(const v8h*)(pp + 8);
      ((v8h*)&pa1)[0] = *(const v8h*)(pp + 32);
      ((v8h*)&pa1)[1] = *(const v8h*)(pp + 40);
    }
    // ---- O += P @ V : 8 WMMAs (Vt[b,h,d,s] contiguous along keys) ----
    #pragma unroll
    for (int hh = 0; hh < 2; ++hh) {
      int kk = t + hh * 32 + ((lane >> 4) << 4);
      if (kk > SLEN - 16) kk = SLEN - 16;           // clamped keys carry p==0
      #pragma unroll
      for (int j = 0; j < 4; ++j) {
        const _Float16* vp = Vt + headoff + (size_t)(j * 16 + col) * SLEN + kk;
        v16h vf;
        ((v8h*)&vf)[0] = *(const v8h*)vp;
        ((v8h*)&vf)[1] = *(const v8h*)(vp + 8);
        o[j] = wmma16(hh ? pa1 : pa0, vf, o[j]);
      }
    }
  }

  // ---- normalize + store ctx[b,s,h*64+d] as f16 ----
  float rl[8];
  #pragma unroll
  for (int r = 0; r < 8; ++r) rl[r] = 1.0f / lrun[r];
  #pragma unroll
  for (int j = 0; j < 4; ++j)
    #pragma unroll
    for (int r = 0; r < 8; ++r) {
      int q = qbase + rowbase + r;
      int d = j * 16 + col;
      ctx[((size_t)b * SLEN + q) * HIDDEN + h * HEADD + d] = (_Float16)(o[j][r] * rl[r]);
    }
}

// -------------------- host launch --------------------
extern "C" void kernel_launch(void* const* d_in, const int* in_sizes, int n_in,
                              void* d_out, int out_size, void* d_ws, size_t ws_size,
                              hipStream_t stream) {
  (void)in_sizes; (void)n_in; (void)out_size; (void)ws_size;
  const float* q  = (const float*)d_in[0];
  const float* k  = (const float*)d_in[1];
  const float* v  = (const float*)d_in[2];
  /* d_in[3] = mask: causal mask applied analytically in attn_kernel */
  const float* Wq = (const float*)d_in[4];
  const float* Wk = (const float*)d_in[5];
  const float* Wv = (const float*)d_in[6];
  const float* Wo = (const float*)d_in[7];
  const float* bo = (const float*)d_in[8];
  float* out = (float*)d_out;

  const size_t nAct = (size_t)MTOK * HIDDEN;   // 4M elems
  const size_t nW   = (size_t)NDIM * KDIM;     // 1M elems
  _Float16* Xq16 = (_Float16*)d_ws;            // also reused as ctx after QKV GEMMs
  _Float16* Xk16 = Xq16 + nAct;
  _Float16* Xv16 = Xk16 + nAct;
  _Float16* Wq16 = Xv16 + nAct;
  _Float16* Wk16 = Wq16 + nW;
  _Float16* Wv16 = Wk16 + nW;
  _Float16* Wo16 = Wv16 + nW;
  _Float16* Q16  = Wo16 + nW;
  _Float16* K16  = Q16 + nAct;
  _Float16* Vt16 = K16 + nAct;                 // total 28M halfs = 56MB of d_ws
  _Float16* ctx  = Xq16;                       // alias: Xq16 dead after Q projection

  // 0) one-pass fp32 -> f16
  dim3 cblk(256);
  cvt_kernel<<<dim3((unsigned)(nAct / 2048)), cblk, 0, stream>>>(q, Xq16, (int)nAct);
  cvt_kernel<<<dim3((unsigned)(nAct / 2048)), cblk, 0, stream>>>(k, Xk16, (int)nAct);
  cvt_kernel<<<dim3((unsigned)(nAct / 2048)), cblk, 0, stream>>>(v, Xv16, (int)nAct);
  cvt_kernel<<<dim3((unsigned)(nW / 2048)),   cblk, 0, stream>>>(Wq, Wq16, (int)nW);
  cvt_kernel<<<dim3((unsigned)(nW / 2048)),   cblk, 0, stream>>>(Wk, Wk16, (int)nW);
  cvt_kernel<<<dim3((unsigned)(nW / 2048)),   cblk, 0, stream>>>(Wv, Wv16, (int)nW);
  cvt_kernel<<<dim3((unsigned)(nW / 2048)),   cblk, 0, stream>>>(Wo, Wo16, (int)nW);

  dim3 ggrid(NDIM / 128, MTOK / 128);   // 8 x 32 workgroups
  dim3 gblk(128);                       // 4 waves, 64x64 tiles each
  const float inv_scale = 0.125f;       // 1/sqrt(64), folded into Q

  gemm_kernel<false><<<ggrid, gblk, 0, stream>>>(Xq16, Wq16, nullptr, Q16, inv_scale, 0);
  gemm_kernel<false><<<ggrid, gblk, 0, stream>>>(Xk16, Wk16, nullptr, K16, 1.0f, 0);
  gemm_kernel<false><<<ggrid, gblk, 0, stream>>>(Xv16, Wv16, nullptr, Vt16, 1.0f, 1);

  attn_kernel<<<dim3((BATCH * HEADS * (SLEN / 16)) / 8), dim3(256), 0, stream>>>(Q16, K16, Vt16, ctx);

  gemm_kernel<true><<<ggrid, gblk, 0, stream>>>(ctx, Wo16, bo, out, 1.0f, 0);
}
